// TriangleMultiplicationOutgoing_52802327937524
// MI455X (gfx1250) — compile-verified, matrix operand order checked by reading, not run on hardware
//
#include <hip/hip_runtime.h>
#include <hip/hip_bf16.h>

// ---------------------------------------------------------------------------
// Types for CDNA5 WMMA (wave32): v_wmma_f32_16x16x32_bf16
//   A: 16x32 bf16  (16 bf16 / lane), B: 32x16 bf16, C/D: 16x16 f32 (8 f32/lane)
// Fragment layout (ISA 7.12.2): lane L holds row M=L%16 (A) / col N=L%16 (B),
// as two contiguous 8-element K chunks at K = 8*(L/16) and K = 16 + 8*(L/16).
// => each chunk is one aligned 16-byte load.
// ---------------------------------------------------------------------------
typedef __bf16 bf16;
typedef __attribute__((ext_vector_type(8)))  bf16  bf16x8;
typedef __attribute__((ext_vector_type(16))) bf16  bf16x16;
typedef __attribute__((ext_vector_type(8)))  float f32x8;

#define NPOS (512 * 512)
#define CZ 128

__device__ __forceinline__ bf16x16 make_frag(const bf16* p0, const bf16* p1) {
    bf16x8 lo = *(const bf16x8*)p0;
    bf16x8 hi = *(const bf16x8*)p1;
    return __builtin_shufflevector(lo, hi, 0, 1, 2, 3, 4, 5, 6, 7,
                                           8, 9, 10, 11, 12, 13, 14, 15);
}

__device__ __forceinline__ float sigmoidf_(float x) {
    return 1.f / (1.f + __expf(-x));
}

// LayerNorm of 32 rows x 128 cols held in LDS, 8 threads per row (wave32
// shuffle reduction), result written as bf16 into a 16B-aligned LDS tile
// with row stride 136 (= 272B, multiple of 16B for ds_load_b128 fragments).
__device__ __forceinline__ void ln32rows(const float src[][129], bf16 dst[][136],
                                         const float* __restrict__ g,
                                         const float* __restrict__ b, int tid) {
    int row = tid >> 3, sub = tid & 7;
    float s = 0.f, s2 = 0.f;
    for (int c = sub; c < 128; c += 8) {
        float v = src[row][c];
        s += v; s2 += v * v;
    }
    s  += __shfl_xor(s, 1);  s  += __shfl_xor(s, 2);  s  += __shfl_xor(s, 4);
    s2 += __shfl_xor(s2, 1); s2 += __shfl_xor(s2, 2); s2 += __shfl_xor(s2, 4);
    float mean = s * 0.0078125f;
    float var  = s2 * 0.0078125f - mean * mean;
    float rstd = rsqrtf(var + 1e-5f);
    for (int c = sub; c < 128; c += 8)
        dst[row][c] = (bf16)((src[row][c] - mean) * rstd * g[c] + b[c]);
}

// ---------------------------------------------------------------------------
// kprep: transpose + convert 6 weight matrices (128x128 f32, [cin][cout]) to
// bf16 [cout][cin] so WMMA B fragments are contiguous-K 16B loads.
// ---------------------------------------------------------------------------
__global__ void __launch_bounds__(256) kprep(const float* __restrict__ g1,
                                             const float* __restrict__ l1,
                                             const float* __restrict__ g2,
                                             const float* __restrict__ l2,
                                             const float* __restrict__ gw,
                                             const float* __restrict__ ow,
                                             bf16* __restrict__ Wt) {
    int m = blockIdx.x;
    const float* src = (m == 0) ? g1 : (m == 1) ? l1 : (m == 2) ? g2
                     : (m == 3) ? l2 : (m == 4) ? gw : ow;
    bf16* dst = Wt + m * 16384;
    for (int idx = threadIdx.x; idx < 16384; idx += 256) {
        int cout = idx >> 7, cin = idx & 127;
        dst[idx] = (bf16)src[cin * 128 + cout];
    }
}

// ---------------------------------------------------------------------------
// k1: LayerNorm + 5 projections. 32 positions / block, 256 threads (8 waves).
// Each wave owns N-tile nt=waveId (16 cols) x 2 M-tiles; K=128 in 4 WMMA steps.
// Writes A_t / B_t channel-major ([c][pos]) bf16 as packed 16B stores, G f32.
// ---------------------------------------------------------------------------
__global__ void __launch_bounds__(256) k1_proj(
    const float* __restrict__ x, const float* __restrict__ mask,
    const float* __restrict__ ln_g, const float* __restrict__ ln_b,
    const bf16* __restrict__ Wt,                       // g1,l1,g2,l2,gate
    const float* __restrict__ bg1, const float* __restrict__ bl1,
    const float* __restrict__ bg2, const float* __restrict__ bl2,
    const float* __restrict__ bg,
    bf16* __restrict__ At, bf16* __restrict__ Bt, float* __restrict__ G) {
    __shared__ __align__(16) float xs[32][129];
    __shared__ __align__(16) bf16  zs[32][136];
    int tid = threadIdx.x;
    int P0  = blockIdx.x * 32;

    for (int idx = tid; idx < 32 * 128; idx += 256) {
        int p = idx >> 7, c = idx & 127;
        xs[p][c] = x[(size_t)(P0 + p) * CZ + c];
    }
    __syncthreads();
    ln32rows(xs, zs, ln_g, ln_b, tid);
    __syncthreads();

    int w = tid >> 5, lane = tid & 31, ln = lane & 15, h = lane >> 4;
    int nt = w;
    const bf16* Wg1 = Wt;
    const bf16* Wl1 = Wt + 16384;
    const bf16* Wg2 = Wt + 2 * 16384;
    const bf16* Wl2 = Wt + 3 * 16384;
    const bf16* Wgt = Wt + 4 * 16384;
    int wr = (nt * 16 + ln) * 128;   // weight row (c_out) base

    for (int mt = 0; mt < 2; ++mt) {
        bf16x16 af[4];
#pragma unroll
        for (int ks = 0; ks < 4; ++ks)
            af[ks] = make_frag(&zs[mt * 16 + ln][ks * 32 + h * 8],
                               &zs[mt * 16 + ln][ks * 32 + 16 + h * 8]);
        f32x8 ag1 = {0,0,0,0,0,0,0,0}, al1 = {0,0,0,0,0,0,0,0};
        f32x8 ag2 = {0,0,0,0,0,0,0,0}, al2 = {0,0,0,0,0,0,0,0};
        f32x8 agg = {0,0,0,0,0,0,0,0};
#pragma unroll
        for (int ks = 0; ks < 4; ++ks) {
            int co = ks * 32 + h * 8;
            bf16x16 bfr;
            bfr = make_frag(Wg1 + wr + co, Wg1 + wr + co + 16);
            ag1 = __builtin_amdgcn_wmma_f32_16x16x32_bf16(false, af[ks], false, bfr,
                                                          (short)0, ag1, false, false);
            bfr = make_frag(Wl1 + wr + co, Wl1 + wr + co + 16);
            al1 = __builtin_amdgcn_wmma_f32_16x16x32_bf16(false, af[ks], false, bfr,
                                                          (short)0, al1, false, false);
            bfr = make_frag(Wg2 + wr + co, Wg2 + wr + co + 16);
            ag2 = __builtin_amdgcn_wmma_f32_16x16x32_bf16(false, af[ks], false, bfr,
                                                          (short)0, ag2, false, false);
            bfr = make_frag(Wl2 + wr + co, Wl2 + wr + co + 16);
            al2 = __builtin_amdgcn_wmma_f32_16x16x32_bf16(false, af[ks], false, bfr,
                                                          (short)0, al2, false, false);
            bfr = make_frag(Wgt + wr + co, Wgt + wr + co + 16);
            agg = __builtin_amdgcn_wmma_f32_16x16x32_bf16(false, af[ks], false, bfr,
                                                          (short)0, agg, false, false);
        }
        int c = nt * 16 + ln;
        bf16x8 pa, pb;
#pragma unroll
        for (int r = 0; r < 8; ++r) {
            int pos = P0 + mt * 16 + 8 * h + r;   // contiguous in r
            float m  = mask[pos];
            float av = sigmoidf_(ag1[r] + bg1[c]) * (al1[r] + bl1[c]) * m;
            float bv = sigmoidf_(ag2[r] + bg2[c]) * (al2[r] + bl2[c]) * m;
            pa[r] = (bf16)av;
            pb[r] = (bf16)bv;
            G[(size_t)pos * CZ + c] = sigmoidf_(agg[r] + bg[c]);
        }
        size_t obase = (size_t)c * NPOS + P0 + mt * 16 + 8 * h;   // 16B aligned
        *(bf16x8*)(At + obase) = pa;
        *(bf16x8*)(Bt + obase) = pb;
    }
}

// ---------------------------------------------------------------------------
// k2: per-channel GEMM V_c = A_c * B_c^T (512x512x512), 128x128 tile / block.
// Double-buffered LDS fed by global_load_async_to_lds_b128 (ASYNCcnt path).
// 8 waves; wave w = one 16-row M strip x 8 N tiles (64 f32 acc VGPRs).
// ---------------------------------------------------------------------------
__global__ void __launch_bounds__(256) k2_einsum(const bf16* __restrict__ At,
                                                 const bf16* __restrict__ Bt,
                                                 float* __restrict__ vals) {
    __shared__ __align__(16) bf16 As[2][128 * 40];   // 32-K tiles, 80B row pitch
    __shared__ __align__(16) bf16 Bs[2][128 * 40];
    int tid = threadIdx.x;
    int c   = blockIdx.y;
    int it  = blockIdx.x >> 2, jt = blockIdx.x & 3;
    const bf16* Ab = At + (size_t)c * NPOS + (size_t)(it * 128) * 512;
    const bf16* Bb = Bt + (size_t)c * NPOS + (size_t)(jt * 128) * 512;

    unsigned aLds = (unsigned)(unsigned long long)(void*)&As[0][0];
    unsigned bLds = (unsigned)(unsigned long long)(void*)&Bs[0][0];
    int r0 = tid >> 2,          q0 = tid & 3;           // chunk tid
    int r1 = (tid + 256) >> 2,  q1 = (tid + 256) & 3;   // chunk tid+256
    unsigned l0 = (unsigned)(r0 * 80 + q0 * 16);
    unsigned l1 = (unsigned)(r1 * 80 + q1 * 16);

    auto issue = [&](int buf, int ks) {
        unsigned v0  = (unsigned)(r0 * 1024 + ks * 64 + q0 * 16);
        unsigned v1  = (unsigned)(r1 * 1024 + ks * 64 + q1 * 16);
        unsigned off = (unsigned)(buf * (128 * 40 * 2));
        asm volatile("global_load_async_to_lds_b128 %0, %1, %2"
                     :: "v"(aLds + off + l0), "v"(v0), "s"(Ab) : "memory");
        asm volatile("global_load_async_to_lds_b128 %0, %1, %2"
                     :: "v"(bLds + off + l0), "v"(v0), "s"(Bb) : "memory");
        asm volatile("global_load_async_to_lds_b128 %0, %1, %2"
                     :: "v"(aLds + off + l1), "v"(v1), "s"(Ab) : "memory");
        asm volatile("global_load_async_to_lds_b128 %0, %1, %2"
                     :: "v"(bLds + off + l1), "v"(v1), "s"(Bb) : "memory");
    };

    int w = tid >> 5, lane = tid & 31, ln = lane & 15, h = lane >> 4;
    f32x8 z8 = {0,0,0,0,0,0,0,0};
    f32x8 acc[8];
#pragma unroll
    for (int nt = 0; nt < 8; ++nt) acc[nt] = z8;

    issue(0, 0);
    for (int ks = 0; ks < 16; ++ks) {
        int buf = ks & 1;
        if (ks < 15) {
            issue(buf ^ 1, ks + 1);
            asm volatile("s_wait_asynccnt 4" ::: "memory");
        } else {
            asm volatile("s_wait_asynccnt 0" ::: "memory");
        }
        __syncthreads();
        const bf16* ab = &As[buf][0];
        const bf16* bb = &Bs[buf][0];
        bf16x16 afr = make_frag(ab + (w * 16 + ln) * 40 + h * 8,
                                ab + (w * 16 + ln) * 40 + 16 + h * 8);
#pragma unroll
        for (int nt = 0; nt < 8; ++nt) {
            bf16x16 bfr = make_frag(bb + (nt * 16 + ln) * 40 + h * 8,
                                    bb + (nt * 16 + ln) * 40 + 16 + h * 8);
            acc[nt] = __builtin_amdgcn_wmma_f32_16x16x32_bf16(false, afr, false, bfr,
                                                              (short)0, acc[nt],
                                                              false, false);
        }
        __syncthreads();
    }

    float* vb = vals + (size_t)c * NPOS;
#pragma unroll
    for (int nt = 0; nt < 8; ++nt) {
        int j = jt * 128 + nt * 16 + ln;
#pragma unroll
        for (int r = 0; r < 8; ++r) {
            int i = it * 128 + w * 16 + 8 * h + r;
            vb[(size_t)i * 512 + j] = acc[nt][r];   // 64B segments across lanes
        }
    }
}

// ---------------------------------------------------------------------------
// k3: LN(vals) @ out_w + bias, * G, * mask. 32 positions / block.
// ---------------------------------------------------------------------------
__global__ void __launch_bounds__(256) k3_out(
    const float* __restrict__ vals, const float* __restrict__ G,
    const float* __restrict__ mask,
    const float* __restrict__ ln_g, const float* __restrict__ ln_b,
    const bf16* __restrict__ Wout, const float* __restrict__ ob,
    float* __restrict__ out) {
    __shared__ __align__(16) float vt[32][129];
    __shared__ __align__(16) bf16  zs[32][136];
    __shared__ __align__(16) float ot[32][128];
    int tid = threadIdx.x;
    int P0  = blockIdx.x * 32;

    for (int idx = tid; idx < 32 * 128; idx += 256) {
        int cc = idx >> 5, p = idx & 31;            // 128B contiguous per channel row
        vt[p][cc] = vals[(size_t)cc * NPOS + P0 + p];
    }
    __syncthreads();
    ln32rows(vt, zs, ln_g, ln_b, tid);
    __syncthreads();

    int w = tid >> 5, lane = tid & 31, ln = lane & 15, h = lane >> 4;
    int nt = w;
    int wr = (nt * 16 + ln) * 128;
    for (int mt = 0; mt < 2; ++mt) {
        f32x8 acc = {0,0,0,0,0,0,0,0};
#pragma unroll
        for (int ks = 0; ks < 4; ++ks) {
            bf16x16 afr = make_frag(&zs[mt * 16 + ln][ks * 32 + h * 8],
                                    &zs[mt * 16 + ln][ks * 32 + 16 + h * 8]);
            int co = ks * 32 + h * 8;
            bf16x16 bfr = make_frag(Wout + wr + co, Wout + wr + co + 16);
            acc = __builtin_amdgcn_wmma_f32_16x16x32_bf16(false, afr, false, bfr,
                                                          (short)0, acc, false, false);
        }
#pragma unroll
        for (int r = 0; r < 8; ++r)
            ot[mt * 16 + 8 * h + r][nt * 16 + ln] = acc[r];
    }
    __syncthreads();

    for (int idx = tid; idx < 32 * 128; idx += 256) {
        int p = idx >> 7, cc = idx & 127;
        int pos = P0 + p;
        out[(size_t)pos * CZ + cc] =
            (ot[p][cc] + ob[cc]) * G[(size_t)pos * CZ + cc] * mask[pos];
    }
}

// ---------------------------------------------------------------------------
extern "C" void kernel_launch(void* const* d_in, const int* in_sizes, int n_in,
                              void* d_out, int out_size, void* d_ws, size_t ws_size,
                              hipStream_t stream) {
    const float* x   = (const float*)d_in[0];
    const float* msk = (const float*)d_in[1];
    const float* lpg = (const float*)d_in[2];
    const float* lpb = (const float*)d_in[3];
    const float* log_ = (const float*)d_in[4];
    const float* lob = (const float*)d_in[5];
    const float* g1w = (const float*)d_in[6];
    const float* g1b = (const float*)d_in[7];
    const float* g2w = (const float*)d_in[8];
    const float* g2b = (const float*)d_in[9];
    const float* l1w = (const float*)d_in[10];
    const float* l1b = (const float*)d_in[11];
    const float* l2w = (const float*)d_in[12];
    const float* l2b = (const float*)d_in[13];
    const float* gw  = (const float*)d_in[14];
    const float* gb  = (const float*)d_in[15];
    const float* ow  = (const float*)d_in[16];
    const float* ob  = (const float*)d_in[17];
    float* out = (float*)d_out;

    // Workspace layout (bytes): Wt 196608 | At 64Mi | Bt 64Mi | G 128Mi | V 128Mi
    char* ws = (char*)d_ws;
    const size_t WT_B = 6 * 16384 * sizeof(bf16);           // 196608
    const size_t AB_B = (size_t)CZ * NPOS * sizeof(bf16);   // 64 MiB each
    const size_t GV_B = (size_t)CZ * NPOS * sizeof(float);  // 128 MiB each
    bf16*  Wt = (bf16*)ws;
    bf16*  At = (bf16*)(ws + WT_B);
    bf16*  Bt = (bf16*)(ws + WT_B + AB_B);
    float* G  = (float*)(ws + WT_B + 2 * AB_B);
    float* V  = (float*)(ws + WT_B + 2 * AB_B + GV_B);
    (void)in_sizes; (void)n_in; (void)out_size; (void)ws_size;

    kprep<<<6, 256, 0, stream>>>(g1w, l1w, g2w, l2w, gw, ow, Wt);
    k1_proj<<<NPOS / 32, 256, 0, stream>>>(x, msk, lpg, lpb, Wt,
                                           g1b, l1b, g2b, l2b, gb, At, Bt, G);
    k2_einsum<<<dim3(16, 128), 256, 0, stream>>>(At, Bt, V);
    k3_out<<<NPOS / 32, 256, 0, stream>>>(V, G, msk, log_, lob,
                                          Wt + 5 * 16384, ob, out);
}
